// MultiHeadAttention_80951543595567
// MI455X (gfx1250) — matrix operand order repaired and checked by reading, not compiled, not run on hardware
//
#include <hip/hip_runtime.h>

// ---------------- types / helpers ----------------
typedef __bf16 v16bf __attribute__((ext_vector_type(16)));
typedef float  v8f   __attribute__((ext_vector_type(8)));
typedef unsigned int uv4 __attribute__((ext_vector_type(4)));

union FragB16 { v16bf v; uv4 q[2]; };

static __device__ __forceinline__ __bf16 f2bf(float f) {
  unsigned u = __builtin_bit_cast(unsigned, f);
  unsigned r = u + 0x7fffu + ((u >> 16) & 1u);   // round-to-nearest-even
  unsigned short h = (unsigned short)(r >> 16);
  return __builtin_bit_cast(__bf16, h);
}

static __device__ __forceinline__ v8f wmma_bf16(v16bf a, v16bf b, v8f c) {
  // D = A(16x32 bf16) x B(32x16 bf16) + C(16x16 f32)
  return __builtin_amdgcn_wmma_f32_16x16x32_bf16(false, a, false, b, (short)0, c,
                                                 false, false);
}

// CDNA5 async copy: 16 bytes per lane, global -> LDS, tracked by ASYNCcnt.
static __device__ __forceinline__ void async_ld16(const __bf16* g, __bf16* l) {
  unsigned lds_off = (unsigned)(unsigned long long)(const void*)l; // addr[31:0]
  asm volatile("global_load_async_to_lds_b128 %0, %1, off"
               :: "v"(lds_off), "v"((unsigned long long)(const void*)g)
               : "memory");
}

static __device__ __forceinline__ FragB16 load_afrag(const __bf16* p) {
  FragB16 a;
  a.q[0] = *(const uv4*)(p);
  a.q[1] = *(const uv4*)(p + 8);
  return a;
}

#define BB 4
#define TT 2048
#define CC 1024
#define HH 16
#define DD 64
#define HD 1024

// ---------------- conversion kernels ----------------
__global__ void cvt_flat(const float* __restrict__ in, __bf16* __restrict__ out, int n) {
  int i = blockIdx.x * blockDim.x + threadIdx.x;
  if (i < n) out[i] = f2bf(in[i]);
}

// [H,C,D] f32 -> [H,D,C] bf16 (so WMMA B-fragments are K-contiguous)
__global__ void cvt_wT(const float* __restrict__ in, __bf16* __restrict__ out) {
  int i = blockIdx.x * 256 + threadIdx.x;           // H*C*D = 1048576
  int h = i >> 16;                                  // C*D = 65536
  int cd = i & 65535;
  int c = cd >> 6;
  int d = cd & 63;
  out[((size_t)h * DD + d) * CC + c] = f2bf(in[i]);
}

// ---------------- shared GEMM inner machinery ----------------
// Block = 256 threads (8 waves), tile M=256 x N=64, K-step 32.
// Each wave owns two 16-row A tiles -> 8 WMMAs per staged B tile.
// B tile (64 cols x 32 k = 4KB) staged in LDS via async copy, double buffered.
// K-loop manually unrolled x2 so even/odd phases use distinct registers/buffers.

// one K-step: load 4 B fragments from LDS, then 8 back-to-back WMMAs
static __device__ __forceinline__ void gemm_step(
    const __bf16* __restrict__ buf, int l16, int koffB,
    const FragB16& a0, const FragB16& a1, v8f acc0[4], v8f acc1[4]) {
  FragB16 bf4[4];
#pragma unroll
  for (int nt = 0; nt < 4; ++nt) {
    const __bf16* bp = buf + (nt * 16 + l16) * 32 + koffB;
    bf4[nt].q[0] = *(const uv4*)(bp);
    bf4[nt].q[1] = *(const uv4*)(bp + 8);
  }
#pragma unroll
  for (int nt = 0; nt < 4; ++nt) acc0[nt] = wmma_bf16(a0.v, bf4[nt].v, acc0[nt]);
#pragma unroll
  for (int nt = 0; nt < 4; ++nt) acc1[nt] = wmma_bf16(a1.v, bf4[nt].v, acc1[nt]);
}

static __device__ __forceinline__ void gemm_tile(
    const __bf16* __restrict__ arow0,  // lane's A row base, tile 0 (k=0)
    const __bf16* __restrict__ arow1,  // lane's A row base, tile 1 (k=0)
    const __bf16* __restrict__ bcols,  // B cols base: row c contiguous over k
    int ldb, int kdim, __bf16* ldsB /*2*2048 elems*/,
    v8f acc0[4], v8f acc1[4]) {
  const int tid  = threadIdx.x;
  const int lane = tid & 31;
  const int l16  = lane & 15;
  const int hi   = lane >> 4;
  const int koffA = hi ? 16 : 0;      // A frag: K 0..15 / 16..31
  const int koffB = hi ? 16 : 0;      // B frag: K 0..15 / 16..31

  // cooperative staging: thread -> (col, k-chunk), 16B each, 256*16B = 4KB
  const int sc = tid >> 2;            // 0..63
  const int sk = (tid & 3) * 8;       // 0,8,16,24
  const __bf16* gstage = bcols + (size_t)sc * ldb + sk;
  __bf16* l0 = ldsB + sc * 32 + sk;
  __bf16* l1 = l0 + 2048;

  async_ld16(gstage, l0);             // stage 0 -> buf0
  FragB16 ae0 = load_afrag(arow0 + koffA);
  FragB16 ae1 = load_afrag(arow1 + koffA);

  const int nit = kdim / 32;          // even for all our K dims
  for (int it = 0; it < nit; it += 2) {
    // ---- even phase: compute buf0 with aEven; prefetch stage it+1 ----
    async_ld16(gstage + (it + 1) * 32, l1);
    FragB16 ao0 = load_afrag(arow0 + (it + 1) * 32 + koffA);
    FragB16 ao1 = load_afrag(arow1 + (it + 1) * 32 + koffA);
    asm volatile("s_wait_asynccnt 0x1" ::: "memory");   // stage it complete
    __syncthreads();
    gemm_step(ldsB, l16, koffB, ae0, ae1, acc0, acc1);
    __syncthreads();
    // ---- odd phase: compute buf1 with aOdd; prefetch stage it+2 ----
    if (it + 2 < nit) {
      async_ld16(gstage + (it + 2) * 32, l0);
      ae0 = load_afrag(arow0 + (it + 2) * 32 + koffA);
      ae1 = load_afrag(arow1 + (it + 2) * 32 + koffA);
      asm volatile("s_wait_asynccnt 0x1" ::: "memory");
    } else {
      asm volatile("s_wait_asynccnt 0x0" ::: "memory");
    }
    __syncthreads();
    gemm_step(ldsB + 2048, l16, koffB, ao0, ao1, acc0, acc1);
    __syncthreads();
  }
}

// ---------------- QKV projection GEMM ----------------
// grid: (T/256, H, 3*B). block: 256 = 8 waves; wave computes 32x64 tile.
__global__ void __launch_bounds__(256) qkv_gemm(
    const __bf16* __restrict__ xb,                  // [B,T,C]
    const __bf16* __restrict__ wqt, const __bf16* __restrict__ wkt,
    const __bf16* __restrict__ wvt,                 // each [H,D,C]
    __bf16* __restrict__ qb, __bf16* __restrict__ kb,  // [B,H,T,D]
    __bf16* __restrict__ vtb) {                     // [B,H,D,T] (transposed!)
  __shared__ __bf16 bsm[4096];
  const int lane = threadIdx.x & 31;
  const int wid  = threadIdx.x >> 5;
  const int t0   = blockIdx.x * 256 + wid * 32;
  const int h    = blockIdx.y;
  const int w    = blockIdx.z / BB;                 // 0=Q 1=K 2=V
  const int b    = blockIdx.z % BB;
  const int l16  = lane & 15;
  const int hi   = lane >> 4;

  const __bf16* wt = (w == 0) ? wqt : (w == 1) ? wkt : wvt;
  const __bf16* wbase = wt + (size_t)h * DD * CC;
  const __bf16* arow0 = xb + ((size_t)b * TT + t0 + l16) * CC;
  const __bf16* arow1 = arow0 + (size_t)16 * CC;

  v8f acc0[4] = {}, acc1[4] = {};
  gemm_tile(arow0, arow1, wbase, CC, CC, bsm, acc0, acc1);

  const size_t bh = (size_t)b * HH + h;
#pragma unroll
  for (int r = 0; r < 2; ++r) {
    const v8f* acc = r ? acc1 : acc0;
    const int tb = t0 + r * 16;
#pragma unroll
    for (int nt = 0; nt < 4; ++nt) {
#pragma unroll
      for (int j = 0; j < 8; ++j) {
        int tr = tb + (hi ? j + 8 : j);
        int d  = nt * 16 + l16;
        __bf16 val = f2bf(acc[nt][j]);
        if (w == 2)      vtb[(bh * DD + d) * TT + tr] = val;
        else if (w == 0) qb[(bh * TT + tr) * DD + d]  = val;
        else             kb[(bh * TT + tr) * DD + d]  = val;
      }
    }
  }
}

// ---------------- flash attention (causal) ----------------
static __device__ __forceinline__ void load_kfrag(const __bf16* base, int koffB,
                                                  FragB16 kf[2][2]) {
#pragma unroll
  for (int kt = 0; kt < 2; ++kt) {
    const __bf16* krow = base + (size_t)kt * 16 * DD;  // lane = key row
    kf[kt][0].q[0] = *(const uv4*)(krow + koffB);
    kf[kt][0].q[1] = *(const uv4*)(krow + koffB + 8);
    kf[kt][1].q[0] = *(const uv4*)(krow + 32 + koffB);
    kf[kt][1].q[1] = *(const uv4*)(krow + 32 + koffB + 8);
  }
}

// grid: (T/64, H, B). block: 128 = 4 waves; each wave owns a 16-query tile.
__global__ void __launch_bounds__(128) attn_fwd(
    const __bf16* __restrict__ qb,   // [B,H,T,D]
    const __bf16* __restrict__ kb,   // [B,H,T,D]
    const __bf16* __restrict__ vtb,  // [B,H,D,T]
    __bf16* __restrict__ ao) {       // [B,T,H*D]
  __shared__ __bf16 pl[4][16 * 32];
  const int lane = threadIdx.x & 31;
  const int wid  = threadIdx.x >> 5;
  const int qt0  = (blockIdx.x * 4 + wid) * 16;
  const int h    = blockIdx.y;
  const int b    = blockIdx.z;
  const size_t bh = (size_t)b * HH + h;
  const int l16   = lane & 15;
  const int hi    = lane >> 4;
  const int koffA = hi ? 16 : 0;
  const int koffB = hi ? 16 : 0;
  __bf16* lds = &pl[wid][0];

  // Q fragments (16x64 split into two 16x32 A-frags), loaded once
  const __bf16* qrow = qb + (bh * TT + qt0 + l16) * DD;
  FragB16 qa0 = load_afrag(qrow + koffA);
  FragB16 qa1 = load_afrag(qrow + 32 + koffA);

  v8f o[4] = {};
  float m[8], l[8];
#pragma unroll
  for (int j = 0; j < 8; ++j) { m[j] = -__builtin_inff(); l[j] = 0.f; }

  const __bf16* kbase = kb + (bh * TT + l16) * DD;   // + s0*DD per chunk
  FragB16 kc[2][2];
  load_kfrag(kbase, koffB, kc);                      // prefetch chunk 0

  const int send = qt0 + 16;                         // causal chunk bound
  for (int s0 = 0; s0 < send; s0 += 32) {
    // V fragments for this chunk: issue early, consumed after softmax
    FragB16 vf[4];
#pragma unroll
    for (int nt = 0; nt < 4; ++nt) {
      const __bf16* vrow = vtb + (bh * DD + nt * 16 + l16) * TT + s0 + koffB;
      vf[nt].q[0] = *(const uv4*)(vrow);
      vf[nt].q[1] = *(const uv4*)(vrow + 8);
    }

    // S = Q Kᵀ (two 16x16 tiles, K-dim = 64 -> 2 WMMAs each)
    v8f st[2];
#pragma unroll
    for (int kt = 0; kt < 2; ++kt) {
      v8f s = {};
      s = wmma_bf16(qa0.v, kc[kt][0].v, s);
      s = wmma_bf16(qa1.v, kc[kt][1].v, s);
      st[kt] = s;
    }

    // prefetch next chunk's K fragments; softmax math below hides the latency
    FragB16 kn[2][2];
    if (s0 + 32 < send) load_kfrag(kbase + (size_t)(s0 + 32) * DD, koffB, kn);

    // scale + causal mask (C-layout: VGPR j -> row j/j+8, lane%16 -> col)
#pragma unroll
    for (int kt = 0; kt < 2; ++kt)
#pragma unroll
      for (int j = 0; j < 8; ++j) {
        int qi = qt0 + (hi ? j + 8 : j);
        int ki = s0 + kt * 16 + l16;
        float v = st[kt][j] * 0.125f;               // 1/sqrt(64)
        st[kt][j] = (ki <= qi) ? v : -__builtin_inff();
      }

    // online softmax (row reductions across 16-lane halves via shfl_xor)
    float alpha[8];
#pragma unroll
    for (int j = 0; j < 8; ++j) {
      float mj = fmaxf(st[0][j], st[1][j]);
      mj = fmaxf(mj, __shfl_xor(mj, 1));
      mj = fmaxf(mj, __shfl_xor(mj, 2));
      mj = fmaxf(mj, __shfl_xor(mj, 4));
      mj = fmaxf(mj, __shfl_xor(mj, 8));
      float mn = fmaxf(m[j], mj);
      alpha[j] = __expf(m[j] - mn);
      m[j] = mn;
      float p0 = __expf(st[0][j] - mn);
      float p1 = __expf(st[1][j] - mn);
      st[0][j] = p0; st[1][j] = p1;
      float rs = p0 + p1;
      rs += __shfl_xor(rs, 1);
      rs += __shfl_xor(rs, 2);
      rs += __shfl_xor(rs, 4);
      rs += __shfl_xor(rs, 8);
      l[j] = l[j] * alpha[j] + rs;
    }
#pragma unroll
    for (int nt = 0; nt < 4; ++nt)
#pragma unroll
      for (int j = 0; j < 8; ++j) o[nt][j] *= alpha[j];

    // transpose P from C-layout to A-layout through LDS (wave-private slice)
#pragma unroll
    for (int kt = 0; kt < 2; ++kt)
#pragma unroll
      for (int j = 0; j < 8; ++j) {
        int row = hi ? j + 8 : j;
        int col = kt * 16 + l16;
        lds[row * 32 + col] = f2bf(st[kt][j]);
      }
    asm volatile("s_wait_dscnt 0" ::: "memory");
    FragB16 pa;
    pa.q[0] = *(const uv4*)(lds + l16 * 32 + koffA);
    pa.q[1] = *(const uv4*)(lds + l16 * 32 + koffA + 8);

    // O += P(16x32) x V(32x64)
#pragma unroll
    for (int nt = 0; nt < 4; ++nt) o[nt] = wmma_bf16(pa.v, vf[nt].v, o[nt]);

#pragma unroll
    for (int kt = 0; kt < 2; ++kt) {
      kc[kt][0] = kn[kt][0];
      kc[kt][1] = kn[kt][1];
    }
  }

  // epilogue: normalize and write concat-head layout [B,T,H*D]
#pragma unroll
  for (int nt = 0; nt < 4; ++nt)
#pragma unroll
    for (int j = 0; j < 8; ++j) {
      int t = qt0 + (hi ? j + 8 : j);
      int d = nt * 16 + l16;
      float val = o[nt][j] / l[j];
      ao[((size_t)b * TT + t) * HD + h * DD + d] = f2bf(val);
    }
}

// ---------------- output projection GEMM + bias ----------------
// out[n,c] = sum_k ao[n,k]*Wp[c,k] + bp[c]; grid (8192/256, 1024/64), 256 thr
__global__ void __launch_bounds__(256) proj_gemm(
    const __bf16* __restrict__ ao,   // [B*T, 1024]
    const __bf16* __restrict__ wpb,  // [1024, 1024] row c contiguous over k
    const float* __restrict__ bp, float* __restrict__ out) {
  __shared__ __bf16 bsm[4096];
  const int lane = threadIdx.x & 31;
  const int wid  = threadIdx.x >> 5;
  const int n0   = blockIdx.x * 256 + wid * 32;
  const int c0   = blockIdx.y * 64;
  const int l16  = lane & 15;
  const int hi   = lane >> 4;

  const __bf16* arow0 = ao + (size_t)(n0 + l16) * HD;
  const __bf16* arow1 = arow0 + (size_t)16 * HD;
  const __bf16* bcols = wpb + (size_t)c0 * HD;

  v8f acc0[4] = {}, acc1[4] = {};
  gemm_tile(arow0, arow1, bcols, HD, HD, bsm, acc0, acc1);

#pragma unroll
  for (int r = 0; r < 2; ++r) {
    const v8f* acc = r ? acc1 : acc0;
    const int nb = n0 + r * 16;
#pragma unroll
    for (int nt = 0; nt < 4; ++nt) {
      int c = c0 + nt * 16 + l16;
      float bias = bp[c];
#pragma unroll
      for (int j = 0; j < 8; ++j) {
        int n = nb + (hi ? j + 8 : j);
        out[(size_t)n * CC + c] = acc[nt][j] + bias;
      }
    }
  }
}

// ---------------- launch ----------------
extern "C" void kernel_launch(void* const* d_in, const int* in_sizes, int n_in,
                              void* d_out, int out_size, void* d_ws, size_t ws_size,
                              hipStream_t stream) {
  (void)in_sizes; (void)n_in; (void)out_size; (void)ws_size;
  const float* x  = (const float*)d_in[0];
  const float* Wq = (const float*)d_in[1];
  const float* Wk = (const float*)d_in[2];
  const float* Wv = (const float*)d_in[3];
  const float* Wp = (const float*)d_in[4];
  const float* bp = (const float*)d_in[5];
  float* out = (float*)d_out;

  const size_t NX = (size_t)BB * TT * CC;   // 8,388,608
  const size_t NW = (size_t)HH * CC * DD;   // 1,048,576
  __bf16* ws  = (__bf16*)d_ws;              // total ~92 MB of workspace in bf16
  __bf16* xb  = ws;
  __bf16* wqt = xb  + NX;
  __bf16* wkt = wqt + NW;
  __bf16* wvt = wkt + NW;
  __bf16* wpb = wvt + NW;
  __bf16* qb  = wpb + NW;
  __bf16* kb  = qb  + NX;
  __bf16* vtb = kb  + NX;
  __bf16* ao  = vtb + NX;

  cvt_flat<<<dim3((unsigned)(NX / 256)), 256, 0, stream>>>(x, xb, (int)NX);
  cvt_wT  <<<dim3(4096), 256, 0, stream>>>(Wq, wqt);
  cvt_wT  <<<dim3(4096), 256, 0, stream>>>(Wk, wkt);
  cvt_wT  <<<dim3(4096), 256, 0, stream>>>(Wv, wvt);
  cvt_flat<<<dim3((unsigned)(NW / 256)), 256, 0, stream>>>(Wp, wpb, (int)NW);

  qkv_gemm<<<dim3(TT / 256, HH, 3 * BB), 256, 0, stream>>>(
      xb, wqt, wkt, wvt, qb, kb, vtb);
  attn_fwd<<<dim3(TT / 64, HH, BB), 128, 0, stream>>>(qb, kb, vtb, ao);
  proj_gemm<<<dim3((BB * TT) / 256, CC / 64), 256, 0, stream>>>(ao, wpb, bp, out);
}
